// BUDL_20761871909204
// MI455X (gfx1250) — matrix operand order repaired and checked by reading, not compiled, not run on hardware
//
#include <hip/hip_runtime.h>

// ---------------------------------------------------------------------------
// Problem constants (B=4, C=64, H=W=256)
// ---------------------------------------------------------------------------
#define Pdim 65536      // H*W pixels per image
#define Cdim 64
#define HIDc 170

typedef __attribute__((ext_vector_type(16))) _Float16 v16h_t;
typedef __attribute__((ext_vector_type(8)))  _Float16 v8h_t;
typedef __attribute__((ext_vector_type(8)))  float    v8f_t;
typedef __attribute__((ext_vector_type(4)))  unsigned int v4u_t;
typedef __attribute__((ext_vector_type(8)))  int      v8i_t;
typedef __attribute__((ext_vector_type(4)))  int      v4i_t;

#if defined(__has_builtin)
#if __has_builtin(__builtin_amdgcn_tensor_load_to_lds) && \
    __has_builtin(__builtin_amdgcn_s_wait_tensorcnt)
#define HAVE_TDM 1
#endif
#endif
#ifndef HAVE_TDM
#define HAVE_TDM 0
#endif

// ---------------------------------------------------------------------------
// WMMA fragment helpers (CDNA5 v_wmma_f32_16x16x32_f16, wave32)
//   A (16x32, MxK): lane m=lane&15, h=lane>>4 holds K = [8h..8h+7] and
//                   [16+8h..16+8h+7] of row m  -> two 16B loads, row-major src
//   B (32x16, KxN): lane n=lane&15, h=lane>>4 holds K = [8h..8h+7] and
//                   [16+8h..16+8h+7] of column n -> two 16B loads from a [n][k] (K-contig) source
//   D (16x16 f32):  elem r -> (m = r + 8*(lane>>4), n = lane&15)
// ---------------------------------------------------------------------------
__device__ __forceinline__ v16h_t frag_a_rm(const _Float16* src, int ld, int lane) {
  int m = lane & 15, h = lane >> 4;
  const _Float16* p = src + (size_t)m * ld + 8 * h;
  v8h_t lo = *(const v8h_t*)(p);
  v8h_t hi = *(const v8h_t*)(p + 16);
  return __builtin_shufflevector(lo, hi, 0,1,2,3,4,5,6,7,8,9,10,11,12,13,14,15);
}

__device__ __forceinline__ v16h_t frag_b_cm(const _Float16* src, int ld, int lane) {
  int n = lane & 15, h = lane >> 4;
  const _Float16* p = src + (size_t)n * ld + 8 * h;
  v8h_t lo = *(const v8h_t*)(p);
  v8h_t hi = *(const v8h_t*)(p + 16);
  return __builtin_shufflevector(lo, hi, 0,1,2,3,4,5,6,7,8,9,10,11,12,13,14,15);
}

__device__ __forceinline__ v8f_t wmma16(v16h_t a, v16h_t b, v8f_t c) {
  return __builtin_amdgcn_wmma_f32_16x16x32_f16(false, a, false, b, (short)0, c,
                                                false, false);
}

#if HAVE_TDM
// ---------------------------------------------------------------------------
// Tensor Data Mover: 2D tile (rows x cols f16 elements) global -> LDS with
// LDS row padding of 8 halves (pad_interval=32 DWORDs, pad_amount=4 DWORDs).
// D# built per CDNA5 ISA section 8 (group0 128b, group1 256b, groups 2/3 zero
// for a 2D tensor).  6-arg builtin (clang-23 / amdgpu-toolchain): trailing
// int32x8 group + cache policy.  Issued by one lane; completion via
// s_wait_tensorcnt.  Requires cols == 64 (pad interval: power-of-two DWORDs).
// ---------------------------------------------------------------------------
__device__ __forceinline__ void tdm_load_2d_pad8(const _Float16* gsrc,
                                                 _Float16* ldst, int rows,
                                                 int cols) {
  unsigned long long ga = (unsigned long long)(size_t)gsrc;
  unsigned int la = (unsigned int)(size_t)ldst;      // LDS byte offset
  v4u_t g0;
  g0.x = 1u;                                         // count=1, user mode
  g0.y = la;                                         // lds_addr
  g0.z = (unsigned int)ga;                           // global_addr[31:0]
  g0.w = (unsigned int)(ga >> 32) | (2u << 30);      // global_addr[56:32]|type=2
  v8i_t g1;
  g1[0] = (1 << 16)        // data_size = 2 bytes
        | (1 << 20)        // pad_enable
        | (4 << 22)        // pad_interval: 32 DWORDs (=128B = 64 halves)
        | (3 << 25);       // pad_amount: 4 DWORDs (=8 halves)
  g1[1] = (cols & 0xffff) << 16;                         // tensor_dim0[15:0]
  g1[2] = ((unsigned)cols >> 16) | ((rows & 0xffff) << 16);  // dim0 hi | dim1 lo
  g1[3] = ((unsigned)rows >> 16) | ((cols & 0xffff) << 16);  // dim1 hi | tile_dim0
  g1[4] = rows & 0xffff;                                 // tile_dim1 (tile_dim2=0)
  g1[5] = cols;                                          // tensor_dim0_stride lo
  g1[6] = 0;                                             // stride0 hi | stride1 lo
  g1[7] = 0;
  v4i_t gz4 = {0, 0, 0, 0};
  v8i_t gz8 = {0, 0, 0, 0, 0, 0, 0, 0};
  __builtin_amdgcn_tensor_load_to_lds(g0, g1, gz4, gz4, gz8, 0);
}
#endif

// ---------------------------------------------------------------------------
// Zero-fill helper (avoid hipMemsetAsync concerns under graph capture)
// ---------------------------------------------------------------------------
__global__ void k_zero(float* __restrict__ p, int n) {
  int i = blockIdx.x * 256 + threadIdx.x;
  if (i < n) p[i] = 0.f;
}

// ---------------------------------------------------------------------------
// Weight pre-conversion: f32 [Cout][Cin] -> f16 [Mp][Kp], zero padded
// ---------------------------------------------------------------------------
__global__ void k_cvt_w(const float* __restrict__ src, _Float16* __restrict__ dst,
                        int Cout, int Cin, int Kp, int total) {
  int i = blockIdx.x * 256 + threadIdx.x;
  if (i >= total) return;
  int m = i / Kp, k = i - m * Kp;
  dst[i] = (_Float16)((m < Cout && k < Cin) ? src[m * Cin + k] : 0.f);
}

// ---------------------------------------------------------------------------
// Channel LayerNorm over C=64 per pixel: f32 in -> f16 out
// ---------------------------------------------------------------------------
__global__ void k_ln(const float* __restrict__ x, const float* __restrict__ w,
                     const float* __restrict__ b, _Float16* __restrict__ y) {
  int p = blockIdx.x * 256 + threadIdx.x;
  float s = 0.f, ss = 0.f;
  for (int c = 0; c < Cdim; ++c) {
    float v = x[(size_t)c * Pdim + p];
    s += v; ss += v * v;
  }
  float mu = s * (1.f / Cdim);
  float var = ss * (1.f / Cdim) - mu * mu;
  float r = rsqrtf(var + 1e-5f);
  for (int c = 0; c < Cdim; ++c) {
    float v = (x[(size_t)c * Pdim + p] - mu) * r * w[c] + b[c];
    y[(size_t)c * Pdim + p] = (_Float16)v;
  }
}

// ---------------------------------------------------------------------------
// Pointwise (1x1) conv as WMMA GEMM: out[o,p] = sum_c W[o,c]*X[c,p] + bias[o]
// W16: f16 [Mp][Kp] pre-padded.  Weight tile -> LDS via TDM (Kp==64) or wide
// copies; X tile (64 pixels) staged transposed into LDS: Xt[p][k].
// RES=true adds f32 residual and writes f32; else writes f16.
// ---------------------------------------------------------------------------
template <bool RES, typename OutT>
__global__ void k_gemm_pw(const _Float16* __restrict__ X, const _Float16* __restrict__ W16,
                          const float* __restrict__ bias, const float* __restrict__ res,
                          OutT* __restrict__ out, int Cout, int Cin, int Kp) {
  extern __shared__ _Float16 lds[];
  const int ldk = Kp + 8;
  const int Mp  = (Cout + 15) & ~15;
  _Float16* Wl = lds;                       // [Mp][ldk]
  _Float16* Xt = lds + (size_t)Mp * ldk;    // [64][ldk]
  const int t  = threadIdx.x;
  const int p0 = blockIdx.x * 64;

  bool tdm_done = false;
#if HAVE_TDM
  if (Kp == 64) {                            // pad interval requires 128B rows
    if (t == 0) tdm_load_2d_pad8(W16, Wl, Mp, Kp);
    tdm_done = true;
  }
#endif
  if (!tdm_done) {
    const int kc8 = Kp >> 3;
    for (int idx = t; idx < Mp * kc8; idx += 256) {
      int m = idx / kc8, kc = idx - m * kc8;
      *(v8h_t*)(Wl + m * ldk + kc * 8) =
          *(const v8h_t*)(W16 + (size_t)m * Kp + kc * 8);
    }
  }
  // X tile, transposed: one b128 global load per thread, scalar LDS scatter
  for (int idx = t; idx < Kp * 8; idx += 256) {
    int k = idx >> 3, pp0 = (idx & 7) * 8;
    v8h_t xv = {};
    if (k < Cin) xv = *(const v8h_t*)(X + (size_t)k * Pdim + p0 + pp0);
#pragma unroll
    for (int j = 0; j < 8; ++j) Xt[(pp0 + j) * ldk + k] = xv[j];
  }
#if HAVE_TDM
  if (tdm_done) __builtin_amdgcn_s_wait_tensorcnt(0);
#endif
  __syncthreads();

  const int wave = t >> 5, lane = t & 31;
  const int njobs = (Mp >> 4) * 4;
  for (int job = wave; job < njobs; job += 8) {
    int mt = job >> 2, nt = job & 3;
    v8f_t acc = {};
    for (int k0 = 0; k0 < Kp; k0 += 32) {
      v16h_t a = frag_a_rm(Wl + (size_t)mt * 16 * ldk + k0, ldk, lane);
      v16h_t b = frag_b_cm(Xt + (size_t)nt * 16 * ldk + k0, ldk, lane);
      acc = wmma16(a, b, acc);
    }
    int n = lane & 15, hi = lane >> 4;
    int pcol = p0 + nt * 16 + n;
    for (int r = 0; r < 8; ++r) {
      int m = mt * 16 + hi * 8 + r;
      if (m < Cout) {
        float v = acc[r] + bias[m];
        if (RES) v += res[(size_t)m * Pdim + pcol];
        out[(size_t)m * Pdim + pcol] = (OutT)v;
      }
    }
  }
}

// ---------------------------------------------------------------------------
// Depthwise 3x3 SAME (zero pad), f16 in/out, f32 math
// ---------------------------------------------------------------------------
__global__ void k_dw3(const _Float16* __restrict__ X, const float* __restrict__ w9,
                      const float* __restrict__ bias, _Float16* __restrict__ out) {
  int idx = blockIdx.x * 256 + threadIdx.x;      // c*Pdim + p
  int p = idx & (Pdim - 1);
  int c = idx >> 16;
  int hy = p >> 8, wx = p & 255;
  const _Float16* base = X + (size_t)c * Pdim;
  if (hy < 254) __builtin_prefetch(base + (hy + 2) * 256 + wx, 0, 1);
  const float* wk = w9 + c * 9;
  float acc = bias[c];
  for (int dy = -1; dy <= 1; ++dy) {
    int yy = hy + dy;
    if ((unsigned)yy >= 256u) continue;
    for (int dx = -1; dx <= 1; ++dx) {
      int xx = wx + dx;
      if ((unsigned)xx >= 256u) continue;
      acc += wk[(dy + 1) * 3 + (dx + 1)] * (float)base[yy * 256 + xx];
    }
  }
  out[idx] = (_Float16)acc;
}

// Depthwise 3x3 on 340 channels fused with GELU gate:
// out[c] = gelu(dw(c)) * dw(c+170), c in [0,170)
__global__ void k_dw3_gate(const _Float16* __restrict__ X, const float* __restrict__ w9,
                           const float* __restrict__ bias, _Float16* __restrict__ out) {
  int idx = blockIdx.x * 256 + threadIdx.x;      // c*Pdim + p, c in [0,170)
  int p = idx & (Pdim - 1);
  int c = idx >> 16;
  int hy = p >> 8, wx = p & 255;
  const _Float16* b1 = X + (size_t)c * Pdim;
  const _Float16* b2 = X + (size_t)(c + HIDc) * Pdim;
  if (hy < 254) {
    __builtin_prefetch(b1 + (hy + 2) * 256 + wx, 0, 1);
    __builtin_prefetch(b2 + (hy + 2) * 256 + wx, 0, 1);
  }
  const float* w1 = w9 + c * 9;
  const float* w2 = w9 + (c + HIDc) * 9;
  float a1 = bias[c], a2 = bias[c + HIDc];
  for (int dy = -1; dy <= 1; ++dy) {
    int yy = hy + dy;
    if ((unsigned)yy >= 256u) continue;
    for (int dx = -1; dx <= 1; ++dx) {
      int xx = wx + dx;
      if ((unsigned)xx >= 256u) continue;
      float t1 = w1[(dy + 1) * 3 + (dx + 1)];
      float t2 = w2[(dy + 1) * 3 + (dx + 1)];
      a1 += t1 * (float)b1[yy * 256 + xx];
      a2 += t2 * (float)b2[yy * 256 + xx];
    }
  }
  float ge = 0.5f * a1 * (1.f + erff(a1 * 0.70710678118654752f));   // exact GELU
  out[idx] = (_Float16)(ge * a2);
}

// ---------------------------------------------------------------------------
// W-branch row norms: rows n=(c,h) over w (contiguous 256). rprod = rq*rk
// ---------------------------------------------------------------------------
__global__ void k_rownorm_w(const _Float16* __restrict__ qkv, float* __restrict__ rprod) {
  int wave = threadIdx.x >> 5, lane = threadIdx.x & 31;
  int n = blockIdx.x * 8 + wave;                 // 2048 blocks -> 16384 rows
  const _Float16* q = qkv + (size_t)n * 256;
  const _Float16* k = qkv + (size_t)Cdim * Pdim + (size_t)n * 256;
  float sq = 0.f, sk = 0.f;
  for (int i = lane; i < 256; i += 32) {
    float a = (float)q[i]; sq += a * a;
    float b = (float)k[i]; sk += b * b;
  }
  for (int off = 16; off > 0; off >>= 1) {
    sq += __shfl_down(sq, off, 32);
    sk += __shfl_down(sk, off, 32);
  }
  if (lane == 0) {
    float rq = 1.f / fmaxf(sqrtf(sq), 1e-12f);
    float rk = 1.f / fmaxf(sqrtf(sk), 1e-12f);
    rprod[n] = rq * rk;
  }
}

// H-branch row norms: per h over (c,w) -> rq[h], rk[h]
__global__ void k_rownorm_h(const _Float16* __restrict__ qkv,
                            float* __restrict__ rq, float* __restrict__ rk) {
  __shared__ float rsq[256], rsk[256];
  int h = blockIdx.x, t = threadIdx.x;
  const _Float16* q = qkv + h * 256;
  const _Float16* k = qkv + (size_t)Cdim * Pdim + h * 256;
  float sq = 0.f, sk = 0.f;
  for (int c = 0; c < Cdim; ++c) {
    float a = (float)q[(size_t)c * Pdim + t]; sq += a * a;
    float b = (float)k[(size_t)c * Pdim + t]; sk += b * b;
  }
  rsq[t] = sq; rsk[t] = sk; __syncthreads();
  for (int s = 128; s > 0; s >>= 1) {
    if (t < s) { rsq[t] += rsq[t + s]; rsk[t] += rsk[t + s]; }
    __syncthreads();
  }
  if (t == 0) {
    rq[h] = 1.f / fmaxf(sqrtf(rsq[0]), 1e-12f);
    rk[h] = 1.f / fmaxf(sqrtf(rsk[0]), 1e-12f);
  }
}

// ---------------------------------------------------------------------------
// W-branch logits: S[w,u] += temp * sum_n q[n,w]*k[n,u]*rprod[n]
// K split over 8 slices (grid = 16 tiles * 8), 64x64 output tile per block.
// Staging: one b128 global load per thread per operand, transposed scatter.
// ---------------------------------------------------------------------------
__global__ void k_logits_w(const _Float16* __restrict__ qkv, const float* __restrict__ rprod,
                           const float* __restrict__ temp, float* __restrict__ S) {
  __shared__ __align__(16) _Float16 Aq[64 * 40];
  __shared__ __align__(16) _Float16 Bk[64 * 40];
  int tile = blockIdx.x & 15, ks = blockIdx.x >> 4;
  int m0 = (tile >> 2) * 64, n0 = (tile & 3) * 64;
  int t = threadIdx.x, wave = t >> 5, lane = t & 31;
  int mt = wave >> 1, ntA = (wave & 1) * 2, ntB = ntA + 1;
  const _Float16* qb = qkv;
  const _Float16* kb = qkv + (size_t)Cdim * Pdim;
  const int kk = t >> 3;                 // 0..31 (k row within chunk)
  const int col0 = (t & 7) * 8;          // 8 contiguous output columns
  v8f_t acc0 = {}, acc1 = {};
  for (int k0 = ks * 2048; k0 < ks * 2048 + 2048; k0 += 32) {
    __syncthreads();
    {
      float rp = rprod[k0 + kk];
      v8h_t qv = *(const v8h_t*)(qb + (size_t)(k0 + kk) * 256 + m0 + col0);
      v8h_t kv = *(const v8h_t*)(kb + (size_t)(k0 + kk) * 256 + n0 + col0);
#pragma unroll
      for (int j = 0; j < 8; ++j) {
        Aq[(col0 + j) * 40 + kk] = (_Float16)((float)qv[j] * rp);
        Bk[(col0 + j) * 40 + kk] = kv[j];
      }
    }
    __syncthreads();
    v16h_t a  = frag_a_rm(Aq + mt * 16 * 40, 40, lane);
    v16h_t b0 = frag_b_cm(Bk + ntA * 16 * 40, 40, lane);
    v16h_t b1 = frag_b_cm(Bk + ntB * 16 * 40, 40, lane);
    acc0 = wmma16(a, b0, acc0);
    acc1 = wmma16(a, b1, acc1);
  }
  float tmp = temp[0];
  int n = lane & 15, hi = lane >> 4;
  for (int r = 0; r < 8; ++r) {
    int m = m0 + mt * 16 + hi * 8 + r;
    atomicAdd(&S[m * 256 + n0 + ntA * 16 + n], acc0[r] * tmp);
    atomicAdd(&S[m * 256 + n0 + ntB * 16 + n], acc1[r] * tmp);
  }
}

// ---------------------------------------------------------------------------
// H-branch logits: S[h,g] += temp*rq[h]*rk[g] * sum_{c,w} q[c,h,w]*k[c,g,w]
// Operand rows are K-contiguous in global memory -> direct global fragments.
// ---------------------------------------------------------------------------
__global__ void k_logits_h(const _Float16* __restrict__ qkv, const float* __restrict__ rq,
                           const float* __restrict__ rk, const float* __restrict__ temp,
                           float* __restrict__ S) {
  int tile = blockIdx.x & 15, ks = blockIdx.x >> 4;
  int m0 = (tile >> 2) * 64, n0 = (tile & 3) * 64;
  int wave = threadIdx.x >> 5, lane = threadIdx.x & 31;
  int mt = wave >> 1, ntA = (wave & 1) * 2, ntB = ntA + 1;
  const _Float16* qb = qkv;
  const _Float16* kb = qkv + (size_t)Cdim * Pdim;
  v8f_t acc0 = {}, acc1 = {};
  for (int kc = ks * 64; kc < ks * 64 + 64; ++kc) {   // 512 chunks total (c,w0)
    int c = kc >> 3, w0 = (kc & 7) * 32;
    const _Float16* ab = qb + (size_t)c * Pdim + (size_t)(m0 + mt * 16) * 256 + w0;
    const _Float16* b0 = kb + (size_t)c * Pdim + (size_t)(n0 + ntA * 16) * 256 + w0;
    const _Float16* b1 = kb + (size_t)c * Pdim + (size_t)(n0 + ntB * 16) * 256 + w0;
    v16h_t a = frag_a_rm(ab, 256, lane);
    acc0 = wmma16(a, frag_b_cm(b0, 256, lane), acc0);
    acc1 = wmma16(a, frag_b_cm(b1, 256, lane), acc1);
  }
  float tmp = temp[0];
  int n = lane & 15, hi = lane >> 4;
  for (int r = 0; r < 8; ++r) {
    int m = m0 + mt * 16 + hi * 8 + r;
    int g0 = n0 + ntA * 16 + n, g1 = n0 + ntB * 16 + n;
    atomicAdd(&S[m * 256 + g0], acc0[r] * tmp * rq[m] * rk[g0]);
    atomicAdd(&S[m * 256 + g1], acc1[r] * tmp * rq[m] * rk[g1]);
  }
}

// ---------------------------------------------------------------------------
// Row softmax (256 wide), f32 in, f16 out; TRANS writes A^T (for W-branch)
// ---------------------------------------------------------------------------
template <bool TRANS>
__global__ void k_softmax(const float* __restrict__ S, _Float16* __restrict__ A) {
  __shared__ float red[256];
  int row = blockIdx.x, t = threadIdx.x;
  float v = S[row * 256 + t];
  red[t] = v; __syncthreads();
  for (int s = 128; s > 0; s >>= 1) {
    if (t < s) red[t] = fmaxf(red[t], red[t + s]);
    __syncthreads();
  }
  float mx = red[0]; __syncthreads();
  float e = expf(v - mx);
  red[t] = e; __syncthreads();
  for (int s = 128; s > 0; s >>= 1) {
    if (t < s) red[t] += red[t + s];
    __syncthreads();
  }
  float r = e / red[0];
  if (TRANS) A[t * 256 + row] = (_Float16)r;
  else       A[row * 256 + t] = (_Float16)r;
}

// ---------------------------------------------------------------------------
// W-branch attn*V: out[n,u] = sum_w v[n,w] * attnT[u,w]   (n=(c,h), 16384 rows)
// Both operands K-contiguous in global memory -> direct fragments.
// ---------------------------------------------------------------------------
__global__ void k_av_w(const _Float16* __restrict__ qkv, const _Float16* __restrict__ At,
                       _Float16* __restrict__ vout) {
  int m0 = (blockIdx.x >> 2) * 64, n0 = (blockIdx.x & 3) * 64;
  int wave = threadIdx.x >> 5, lane = threadIdx.x & 31;
  int mt = wave >> 1, ntA = (wave & 1) * 2, ntB = ntA + 1;
  const _Float16* vb = qkv + (size_t)2 * Cdim * Pdim;   // v rows: n*256 contiguous
  v8f_t acc0 = {}, acc1 = {};
  for (int k0 = 0; k0 < 256; k0 += 32) {
    v16h_t a = frag_a_rm(vb + (size_t)(m0 + mt * 16) * 256 + k0, 256, lane);
    acc0 = wmma16(a, frag_b_cm(At + (size_t)(n0 + ntA * 16) * 256 + k0, 256, lane), acc0);
    acc1 = wmma16(a, frag_b_cm(At + (size_t)(n0 + ntB * 16) * 256 + k0, 256, lane), acc1);
  }
  int n = lane & 15, hi = lane >> 4;
  for (int r = 0; r < 8; ++r) {
    int m = m0 + mt * 16 + hi * 8 + r;
    vout[(size_t)m * 256 + n0 + ntA * 16 + n] = (_Float16)acc0[r];
    vout[(size_t)m * 256 + n0 + ntB * 16 + n] = (_Float16)acc1[r];
  }
}

// ---------------------------------------------------------------------------
// H-branch attn*V: out[c,h,w] = sum_g attn[h,g] * v[c,g,w]
// A=attn direct (K-contig); V tiles transposed through LDS (b128 global side).
// ---------------------------------------------------------------------------
__global__ void k_av_h(const _Float16* __restrict__ qkv, const _Float16* __restrict__ Attn,
                       _Float16* __restrict__ vout) {
  __shared__ __align__(16) _Float16 Bt[64 * 40];
  int c = blockIdx.x >> 4, tile = blockIdx.x & 15;
  int m0 = (tile >> 2) * 64, n0 = (tile & 3) * 64;
  const _Float16* vb = qkv + (size_t)2 * Cdim * Pdim + (size_t)c * Pdim;
  int t = threadIdx.x, wave = t >> 5, lane = t & 31;
  int mt = wave >> 1, ntA = (wave & 1) * 2, ntB = ntA + 1;
  const int gg = t >> 3, col0 = (t & 7) * 8;
  v8f_t acc0 = {}, acc1 = {};
  for (int g0 = 0; g0 < 256; g0 += 32) {
    __syncthreads();
    {
      v8h_t vv = *(const v8h_t*)(vb + (size_t)(g0 + gg) * 256 + n0 + col0);
#pragma unroll
      for (int j = 0; j < 8; ++j) Bt[(col0 + j) * 40 + gg] = vv[j];
    }
    __syncthreads();
    v16h_t a = frag_a_rm(Attn + (size_t)(m0 + mt * 16) * 256 + g0, 256, lane);
    acc0 = wmma16(a, frag_b_cm(Bt + ntA * 16 * 40, 40, lane), acc0);
    acc1 = wmma16(a, frag_b_cm(Bt + ntB * 16 * 40, 40, lane), acc1);
  }
  int n = lane & 15, hi = lane >> 4;
  for (int r = 0; r < 8; ++r) {
    int h = m0 + mt * 16 + hi * 8 + r;
    vout[(size_t)c * Pdim + h * 256 + n0 + ntA * 16 + n] = (_Float16)acc0[r];
    vout[(size_t)c * Pdim + h * 256 + n0 + ntB * 16 + n] = (_Float16)acc1[r];
  }
}

// ---------------------------------------------------------------------------
// Host side
// ---------------------------------------------------------------------------
struct AttnP {
  const float *ln_w, *ln_b, *qkv_b, *dw_w, *dw_b, *proj_b, *temp;
  const _Float16 *qkv16, *proj16;
};

extern "C" void kernel_launch(void* const* d_in, const int* in_sizes, int n_in,
                              void* d_out, int out_size, void* d_ws, size_t ws_size,
                              hipStream_t stream) {
  (void)in_sizes; (void)n_in; (void)out_size; (void)ws_size;
  const float* x = (const float*)d_in[0];
  const float* n2_w      = (const float*)d_in[20];
  const float* n2_b      = (const float*)d_in[21];
  const float* ffn_in_w  = (const float*)d_in[22];
  const float* ffn_in_b  = (const float*)d_in[23];
  const float* ffn_dw_w  = (const float*)d_in[24];
  const float* ffn_dw_b  = (const float*)d_in[25];
  const float* ffn_out_w = (const float*)d_in[26];
  const float* ffn_out_b = (const float*)d_in[27];
  float* outp = (float*)d_out;

  // Per-batch workspace layout (bytes, reused across the batch loop)
  char* ws = (char*)d_ws;
  _Float16* qkvA = (_Float16*)(ws + 0);                 // 192*P f16 = 25165824 B
  _Float16* qkvB = (_Float16*)(ws + 25165824);          // 192*P f16
  _Float16* lnY  = (_Float16*)(ws + 50331648);          // 64*P  f16 =  8388608 B
  _Float16* vout = (_Float16*)(ws + 58720256);          // 64*P  f16
  float*    mbuf = (float*)   (ws + 67108864);          // 64*P  f32 = 16777216 B
  float*    zbuf = (float*)   (ws + 83886080);          // 64*P  f32
  float*    S    = (float*)   (ws + 100663296);         // 256*256 f32
  _Float16* At   = (_Float16*)(ws + 100925440);         // 256*256 f16
  float*    rpr  = (float*)   (ws + 101056512);         // 16384 f32
  float*    rqh  = (float*)   (ws + 101122048);         // 256 f32
  float*    rkh  = (float*)   (ws + 101123072);         // 256 f32
  _Float16* wq16 = (_Float16*)(ws + 101124096);         // 192x64
  _Float16* wp16 = (_Float16*)(ws + 101148672);         // 64x64
  _Float16* hq16 = (_Float16*)(ws + 101156864);         // 192x64
  _Float16* hp16 = (_Float16*)(ws + 101181440);         // 64x64
  _Float16* fi16 = (_Float16*)(ws + 101189632);         // 352x64
  _Float16* fo16 = (_Float16*)(ws + 101234688);         // 64x192
  _Float16* hpre = qkvA;                                // 340*P f16 spans qkvA+qkvB
  _Float16* gbuf = vout;                                // 170*P f16 spans vout+mbuf

  AttnP wp = { (const float*)d_in[2],  (const float*)d_in[3],  (const float*)d_in[5],
               (const float*)d_in[6],  (const float*)d_in[7],  (const float*)d_in[9],
               (const float*)d_in[10], wq16, wp16 };
  AttnP hp = { (const float*)d_in[11], (const float*)d_in[12], (const float*)d_in[14],
               (const float*)d_in[15], (const float*)d_in[16], (const float*)d_in[18],
               (const float*)d_in[19], hq16, hp16 };

  // One-time (per call) weight conversion to padded f16
  auto cvt = [&](const float* src, _Float16* dst, int Cout, int Cin, int Kp) {
    int Mp = (Cout + 15) & ~15;
    int total = Mp * Kp;
    k_cvt_w<<<(total + 255) / 256, 256, 0, stream>>>(src, dst, Cout, Cin, Kp, total);
  };
  cvt((const float*)d_in[4],  wq16, 192, 64, 64);
  cvt((const float*)d_in[8],  wp16, 64,  64, 64);
  cvt((const float*)d_in[13], hq16, 192, 64, 64);
  cvt((const float*)d_in[17], hp16, 64,  64, 64);
  cvt(ffn_in_w,  fi16, 340, 64, 64);
  cvt(ffn_out_w, fo16, 64, 170, 192);

  const size_t shm_qkv  = (size_t)(192 + 64) * (64 + 8) * 2;   // 36864 B
  const size_t shm_proj = (size_t)(64 + 64)  * (64 + 8) * 2;   // 18432 B
  const size_t shm_fin  = (size_t)(352 + 64) * (64 + 8) * 2;   // 59904 B
  const size_t shm_fout = (size_t)(64 + 64)  * (192 + 8) * 2;  // 51200 B

  auto run_attn = [&](const float* in, float* outBuf, const AttnP& P, bool hbranch) {
    k_ln<<<Pdim / 256, 256, 0, stream>>>(in, P.ln_w, P.ln_b, lnY);
    k_gemm_pw<false, _Float16><<<Pdim / 64, 256, shm_qkv, stream>>>(
        lnY, P.qkv16, P.qkv_b, (const float*)nullptr, qkvA, 192, 64, 64);
    k_dw3<<<192 * Pdim / 256, 256, 0, stream>>>(qkvA, P.dw_w, P.dw_b, qkvB);
    k_zero<<<(65536 + 255) / 256, 256, 0, stream>>>(S, 65536);
    if (!hbranch) {
      k_rownorm_w<<<2048, 256, 0, stream>>>(qkvB, rpr);
      k_logits_w<<<128, 256, 0, stream>>>(qkvB, rpr, P.temp, S);
      k_softmax<true><<<256, 256, 0, stream>>>(S, At);
      k_av_w<<<1024, 256, 0, stream>>>(qkvB, At, vout);
    } else {
      k_rownorm_h<<<256, 256, 0, stream>>>(qkvB, rqh, rkh);
      k_logits_h<<<128, 256, 0, stream>>>(qkvB, rqh, rkh, P.temp, S);
      k_softmax<false><<<256, 256, 0, stream>>>(S, At);
      k_av_h<<<1024, 256, 0, stream>>>(qkvB, At, vout);
    }
    k_gemm_pw<true, float><<<Pdim / 64, 256, shm_proj, stream>>>(
        vout, P.proj16, P.proj_b, in, outBuf, 64, 64, 64);
  };

  for (int b = 0; b < 4; ++b) {
    const float* xb   = x    + (size_t)b * Cdim * Pdim;
    float*       outb = outp + (size_t)b * Cdim * Pdim;

    run_attn(xb,   mbuf, wp, false);     // W x W attention  -> m
    run_attn(mbuf, zbuf, hp, true);      // H x H attention  -> z

    // FFN: out = z + conv_out( gelu(x1)*x2 ) with [x1,x2]=dw3(conv_in(LN(z)))
    k_ln<<<Pdim / 256, 256, 0, stream>>>(zbuf, n2_w, n2_b, lnY);
    k_gemm_pw<false, _Float16><<<Pdim / 64, 256, shm_fin, stream>>>(
        lnY, fi16, ffn_in_b, (const float*)nullptr, hpre, 340, 64, 64);
    k_dw3_gate<<<HIDc * Pdim / 256, 256, 0, stream>>>(hpre, ffn_dw_w, ffn_dw_b, gbuf);
    k_gemm_pw<true, float><<<Pdim / 64, 256, shm_fout, stream>>>(
        gbuf, fo16, ffn_out_b, zbuf, outb, 64, 170, 192);
  }
}
